// Point_Transformer_cls_7662221656509
// MI455X (gfx1250) — matrix operand order repaired and checked by reading, not compiled, not run on hardware
//
#include <hip/hip_runtime.h>
#include <math.h>

// CDNA5 (gfx1250) wave32 WMMA types
typedef __attribute__((ext_vector_type(16))) __bf16 v16bf;
typedef __attribute__((ext_vector_type(8)))  __bf16 v8bf;
typedef __attribute__((ext_vector_type(8)))  float  v8f;
typedef __attribute__((ext_vector_type(4)))  float  v4f;

__device__ __forceinline__ v8f wmma_bf16(v16bf a, v16bf b, v8f c) {
  return __builtin_amdgcn_wmma_f32_16x16x32_bf16(false, a, false, b, (short)0, c,
                                                 false, false);
}

// A-operand (16-bit, 16x32): lane L holds row M=L%16; this code's K order: lanes 0-15 K=0..15,
// lanes 16-31 K=16..31 -> one contiguous 32-byte load at +(lane&16).
__device__ __forceinline__ v16bf loadA(const __bf16* p) {
  return *(const v16bf*)p;
}
// B-operand (16-bit, 32x16): lane L holds col N=L%16; this code's K order: lanes 0-15 K=0..15,
// lanes 16-31 K=16..31 -> one contiguous 32-byte load at +(lane&16).
__device__ __forceinline__ v16bf loadB(const __bf16* p) {
  return *(const v16bf*)p;
}

// Reductions across the 16-lane half that holds one D-tile row.
__device__ __forceinline__ float redmax16(float v) {
  v = fmaxf(v, __shfl_xor(v, 1, 32));
  v = fmaxf(v, __shfl_xor(v, 2, 32));
  v = fmaxf(v, __shfl_xor(v, 4, 32));
  v = fmaxf(v, __shfl_xor(v, 8, 32));
  return v;
}
__device__ __forceinline__ float redsum16(float v) {
  v += __shfl_xor(v, 1, 32);
  v += __shfl_xor(v, 2, 32);
  v += __shfl_xor(v, 4, 32);
  v += __shfl_xor(v, 8, 32);
  return v;
}

#define NPTS 2048
#define BATCH 16

// ---------------------------------------------------------------------------
// Conv stem. LDS: w2 (bf16, [o][k]) 32 KB + h0 ([n][c], K-contiguous) 32 KB.
// ---------------------------------------------------------------------------
__global__ __launch_bounds__(256)
void k_stem(const float* __restrict__ x, const float* __restrict__ w1,
            const float* __restrict__ g1, const float* __restrict__ b1,
            const float* __restrict__ w2, const float* __restrict__ g2,
            const float* __restrict__ b2,
            float* __restrict__ h, __bf16* __restrict__ hbf) {
  __shared__ __bf16 w2s[128 * 128];  // [o][k]
  __shared__ __bf16 h0t[128 * 128];  // [n][c]
  const int lane = threadIdx.x & 31;
  const int wave = threadIdx.x >> 5;
  const int b  = blockIdx.x >> 4;
  const int n0 = (blockIdx.x & 15) * 128;

  for (int i2 = threadIdx.x; i2 < 128 * 128 / 8; i2 += 256) {
    int lin = i2 * 8;
    v4f a = *(const v4f*)&w2[lin];
    v4f c = *(const v4f*)&w2[lin + 4];
    v8bf o;
#pragma unroll
    for (int e = 0; e < 4; ++e) { o[e] = (__bf16)a[e]; o[e + 4] = (__bf16)c[e]; }
    *(v8bf*)&w2s[lin] = o;
  }
  for (int idx = threadIdx.x; idx < 128 * 128; idx += 256) {
    int c = idx >> 7, n = idx & 127;
    float x0 = x[((size_t)b * 3 + 0) * NPTS + n0 + n];
    float x1 = x[((size_t)b * 3 + 1) * NPTS + n0 + n];
    float x2 = x[((size_t)b * 3 + 2) * NPTS + n0 + n];
    float s = w1[c * 3 + 0] * x0 + w1[c * 3 + 1] * x1 + w1[c * 3 + 2] * x2;
    s = fmaxf(g1[c] * s + b1[c], 0.0f);
    h0t[n * 128 + c] = (__bf16)s;
  }
  __syncthreads();

  const v8f vzero = {};
  v8f acc[8];
#pragma unroll
  for (int t = 0; t < 8; ++t) acc[t] = vzero;

#pragma unroll
  for (int kc = 0; kc < 4; ++kc) {
    v16bf a = loadA(&w2s[(wave * 16 + (lane & 15)) * 128 + kc * 32 +
                         (lane & 16)]);
#pragma unroll
    for (int ns = 0; ns < 8; ++ns) {
      v16bf bb = loadB(&h0t[(ns * 16 + (lane & 15)) * 128 + kc * 32 +
                            (lane & 16)]);
      acc[ns] = wmma_bf16(a, bb, acc[ns]);
    }
  }
#pragma unroll
  for (int ns = 0; ns < 8; ++ns) {
#pragma unroll
    for (int r = 0; r < 8; ++r) {
      int o = wave * 16 + r + ((lane & 16) >> 1);
      int n = n0 + ns * 16 + (lane & 15);
      float val = fmaxf(g2[o] * acc[ns][r] + b2[o], 0.0f);
      size_t gi = ((size_t)b * 128 + o) * NPTS + n;
      h[gi] = val;
      hbf[gi] = (__bf16)val;
    }
  }
}

// ---------------------------------------------------------------------------
// Generic channel matmul. LDS: X tile [n][C] 32 KB (K-contiguous) + W slab
// [64][<=512] bf16 64 KB (K staged in <=2 phases for C=1024).
// ---------------------------------------------------------------------------
__global__ __launch_bounds__(256)
void k_chanmm(const float* __restrict__ W, const __bf16* __restrict__ X,
              const float* __restrict__ bias, const float* __restrict__ gamma,
              const float* __restrict__ beta,
              float* __restrict__ outf, __bf16* __restrict__ outb,
              int O, int Cc, int Nn, int nTiles, int act, int out_nmajor) {
  __shared__ __bf16 Xs[16384];   // [j][Cc], Cc*nw <= 16384 for all call sites
  __shared__ __bf16 Wsh[32768];  // [ol][CH], CH = min(Cc,512)
  const int lane = threadIdx.x & 31;
  const int wave = threadIdx.x >> 5;
  const int b  = blockIdx.x / nTiles;
  const int nt = blockIdx.x % nTiles;
  const int n0 = nt * 128;
  const int nw = (Nn - n0 < 128) ? (Nn - n0) : 128;
  const int obase = blockIdx.y * 64;

  // stage X tile transposed: Xs[j*Cc + k]
  const int tot = Cc * nw;
  for (int i2 = threadIdx.x; i2 < tot / 8; i2 += 256) {
    int lin = i2 * 8;
    int k = lin / nw, j0 = lin - k * nw;  // nw is a multiple of 8
    v8bf val = *(const v8bf*)&X[((size_t)b * Cc + k) * Nn + n0 + j0];
#pragma unroll
    for (int e = 0; e < 8; ++e) Xs[(j0 + e) * Cc + k] = val[e];
  }

  const int osub = wave >> 1;
  const v8f vzero = {};
  v8f acc[4];
#pragma unroll
  for (int t = 0; t < 4; ++t) acc[t] = vzero;

  const int CH = (Cc > 512) ? 512 : Cc;
  const int phases = Cc / CH;
  for (int ph = 0; ph < phases; ++ph) {
    __syncthreads();
    for (int i2 = threadIdx.x; i2 < (64 * CH) / 8; i2 += 256) {
      int lin = i2 * 8;
      int ol = lin / CH, kk = lin - ol * CH;
      int oc = obase + ol; if (oc > O - 1) oc = O - 1;
      const float* wp = &W[(size_t)oc * Cc + ph * CH + kk];
      v4f a = *(const v4f*)wp;
      v4f c = *(const v4f*)(wp + 4);
      v8bf o;
#pragma unroll
      for (int e = 0; e < 4; ++e) { o[e] = (__bf16)a[e]; o[e + 4] = (__bf16)c[e]; }
      *(v8bf*)&Wsh[lin] = o;
    }
    __syncthreads();

    const int kchunks = CH >> 5;
    for (int kc = 0; kc < kchunks; ++kc) {
      v16bf a = loadA(&Wsh[(osub * 16 + (lane & 15)) * CH + kc * 32 +
                           (lane & 16)]);
#pragma unroll
      for (int t = 0; t < 4; ++t) {
        int ns = (wave & 1) + 2 * t;  // wave-uniform: EXEC stays full for WMMA
        if (ns * 16 < nw) {
          v16bf bb = loadB(&Xs[(ns * 16 + (lane & 15)) * Cc + ph * CH +
                               kc * 32 + (lane & 16)]);
          acc[t] = wmma_bf16(a, bb, acc[t]);
        }
      }
    }
  }

#pragma unroll
  for (int t = 0; t < 4; ++t) {
    int ns = (wave & 1) + 2 * t;
    if (ns * 16 >= nw) continue;
#pragma unroll
    for (int r = 0; r < 8; ++r) {
      int o = obase + osub * 16 + r + ((lane & 16) >> 1);
      int n = n0 + ns * 16 + (lane & 15);
      if (o < O) {
        float val = acc[t][r];
        if (bias)  val += bias[o];
        if (gamma) val = gamma[o] * val + (beta ? beta[o] : 0.0f);
        if (act == 1) val = fmaxf(val, 0.0f);
        else if (act == 2) val = (val > 0.0f) ? val : 0.2f * val;
        if (outf) {
          size_t oi = out_nmajor ? (((size_t)b * Nn + n) * O + o)
                                 : (((size_t)b * O + o) * Nn + n);
          outf[oi] = val;
        }
        if (outb) outb[((size_t)b * O + o) * Nn + n] = (__bf16)val;
      }
    }
  }
}

// ---------------------------------------------------------------------------
// Attention pass 1: flash softmax stats. Per-LANE online (max, sumexp) in the
// hot loop (no cross-lane ops); the 16-lane butterfly runs once per row at
// the end with the standard rescale. Stores rmax and 1/rsum.
// ---------------------------------------------------------------------------
__global__ __launch_bounds__(256)
void k_attn_stats(const __bf16* __restrict__ qbf,
                  float* __restrict__ rmax, float* __restrict__ rsinv) {
  __shared__ __bf16 qs[NPTS * 32];  // [n][d], 128 KB
  const int lane = threadIdx.x & 31;
  const int wave = threadIdx.x >> 5;
  const int b = blockIdx.x >> 4;
  const int rbase = (blockIdx.x & 15) * 128;

  for (int i2 = threadIdx.x; i2 < (32 * NPTS) / 8; i2 += 256) {
    int lin = i2 * 8;
    int d = lin >> 11, n = lin & 2047;
    v8bf val = *(const v8bf*)&qbf[(size_t)b * 32 * NPTS + lin];
#pragma unroll
    for (int e = 0; e < 8; ++e) qs[(n + e) * 32 + d] = val[e];
  }
  __syncthreads();

  const int nbase = rbase + wave * 16;
  v16bf a = loadA(&qs[(nbase + (lane & 15)) * 32 + (lane & 16)]);

  float lm[8], ls[8];  // per-lane online stats over this lane's columns
#pragma unroll
  for (int r = 0; r < 8; ++r) { lm[r] = -1e30f; ls[r] = 0.0f; }
  const v8f vzero = {};

  for (int jt = 0; jt < 128; ++jt) {
    v16bf bb = loadB(&qs[(jt * 16 + (lane & 15)) * 32 + (lane & 16)]);
    v8f E = wmma_bf16(a, bb, vzero);
#pragma unroll
    for (int r = 0; r < 8; ++r) {
      float v = E[r];
      float nm = fmaxf(lm[r], v);
      ls[r] = ls[r] * __expf(lm[r] - nm) + __expf(v - nm);
      lm[r] = nm;
    }
  }
  // one butterfly per row: combine the 16 per-lane results
  float fm[8], fs[8];
#pragma unroll
  for (int r = 0; r < 8; ++r) {
    fm[r] = redmax16(lm[r]);
    fs[r] = redsum16(ls[r] * __expf(lm[r] - fm[r]));
  }
  if ((lane & 15) == 0) {
#pragma unroll
    for (int r = 0; r < 8; ++r) {
      int n = nbase + r + ((lane & 16) >> 1);
      rmax[b * NPTS + n] = fm[r];
      rsinv[b * NPTS + n] = 1.0f / fs[r];
    }
  }
}

// ---------------------------------------------------------------------------
// Attention pass 2: per-WG 128-col x_r tile, streaming keys in 32-chunks.
// All tiles stored K-contiguous; att written as one v8bf store per lane.
// ---------------------------------------------------------------------------
__global__ __launch_bounds__(256)
void k_attn_apply(const __bf16* __restrict__ qbf, const __bf16* __restrict__ vbf,
                  const float* __restrict__ rmax, const float* __restrict__ rsinv,
                  const float* __restrict__ h, __bf16* __restrict__ hmx) {
  __shared__ __bf16 qmt[128 * 32];   // [m][d]
  __shared__ __bf16 qnt[32 * 32];    // [n][d]
  __shared__ __bf16 attt[128 * 32];  // [m][n]
  __shared__ __bf16 vch[128 * 32];   // [c][n]
  __shared__ float  colsum[128];

  const int lane = threadIdx.x & 31;
  const int wave = threadIdx.x >> 5;
  const int b  = blockIdx.x >> 4;
  const int m0 = (blockIdx.x & 15) * 128;

  if (threadIdx.x < 128) colsum[threadIdx.x] = 0.0f;
  for (int i2 = threadIdx.x; i2 < (32 * 128) / 8; i2 += 256) {
    int lin = i2 * 8;
    int d = lin >> 7, mj = lin & 127;
    v8bf val = *(const v8bf*)&qbf[(size_t)b * 32 * NPTS + d * NPTS + m0 + mj];
#pragma unroll
    for (int e = 0; e < 8; ++e) qmt[(mj + e) * 32 + d] = val[e];
  }
  __syncthreads();

  const v8f vzero = {};
  v8f acc[8];
#pragma unroll
  for (int t = 0; t < 8; ++t) acc[t] = vzero;

  for (int n0 = 0; n0 < NPTS; n0 += 32) {
    for (int i2 = threadIdx.x; i2 < (32 * 32) / 8; i2 += 256) {
      int lin = i2 * 8;
      int d = lin >> 5, nn = lin & 31;
      v8bf val = *(const v8bf*)&qbf[(size_t)b * 32 * NPTS + d * NPTS + n0 + nn];
#pragma unroll
      for (int e = 0; e < 8; ++e) qnt[(nn + e) * 32 + d] = val[e];
    }
    for (int i2 = threadIdx.x; i2 < (128 * 32) / 8; i2 += 256) {
      int lin = i2 * 8;  // vch already K(n)-contiguous
      *(v8bf*)&vch[lin] =
          *(const v8bf*)&vbf[((size_t)b * 128 + (lin >> 5)) * NPTS + n0 + (lin & 31)];
    }
    __syncthreads();

    {  // energy tiles: 2 n-subtiles x 8 m-subtiles over 8 waves (2 each)
      const int nsub = wave & 1;
      v16bf a = loadA(&qnt[(nsub * 16 + (lane & 15)) * 32 + (lane & 16)]);
      const int nlb = nsub * 16 + ((lane & 16) >> 1);
#pragma unroll
      for (int t = 0; t < 2; ++t) {
        const int msub = (wave >> 1) + t * 4;
        const int mloc = msub * 16 + (lane & 15);
        v16bf bb = loadB(&qmt[mloc * 32 + (lane & 16)]);
        v8f E = wmma_bf16(a, bb, vzero);
        float cp = 0.0f;
        v8bf pv;
#pragma unroll
        for (int r = 0; r < 8; ++r) {
          int ng = n0 + nlb + r;
          float p = __expf(E[r] - rmax[b * NPTS + ng]) * rsinv[b * NPTS + ng];
          pv[r] = (__bf16)p;
          cp += p;
        }
        *(v8bf*)&attt[mloc * 32 + nlb] = pv;   // one ds_store_b128
        atomicAdd(&colsum[mloc], cp);          // ds_add_f32
      }
    }
    __syncthreads();

    {  // x_r accumulation: v[c,n-chunk] @ att[n-chunk,m]
      v16bf a = loadA(&vch[(wave * 16 + (lane & 15)) * 32 + (lane & 16)]);
#pragma unroll
      for (int msub = 0; msub < 8; ++msub) {
        v16bf bb = loadB(&attt[(msub * 16 + (lane & 15)) * 32 + (lane & 16)]);
        acc[msub] = wmma_bf16(a, bb, acc[msub]);
      }
    }
    __syncthreads();
  }

#pragma unroll
  for (int msub = 0; msub < 8; ++msub) {
    int mloc = msub * 16 + (lane & 15);
    float inv = 1.0f / (1e-9f + colsum[mloc]);
#pragma unroll
    for (int r = 0; r < 8; ++r) {
      int c = wave * 16 + r + ((lane & 16) >> 1);
      size_t gi = ((size_t)b * 128 + c) * NPTS + m0 + mloc;
      hmx[gi] = (__bf16)(h[gi] - acc[msub][r] * inv);
    }
  }
}

// ---------------------------------------------------------------------------
// Residual update: h += relu(g*t + b); refresh bf16 copy; scatter into concat.
// ---------------------------------------------------------------------------
__global__ __launch_bounds__(256)
void k_resid(const float* __restrict__ t, const float* __restrict__ gg,
             const float* __restrict__ bb, float* __restrict__ h,
             __bf16* __restrict__ hbf, __bf16* __restrict__ cat, int slot) {
  size_t idx = (size_t)blockIdx.x * 256 + threadIdx.x;  // B*128*2048 total
  int n = (int)(idx & 2047);
  int c = (int)((idx >> 11) & 127);
  int b = (int)(idx >> 18);
  float val = h[idx] + fmaxf(gg[c] * t[idx] + bb[c], 0.0f);
  h[idx] = val;
  __bf16 bv = (__bf16)val;
  hbf[idx] = bv;
  cat[((size_t)b * 512 + slot * 128 + c) * NPTS + n] = bv;
}

// ---------------------------------------------------------------------------
// Fuse conv (1024x512) + affine + LeakyReLU + max-pool over N. Per-lane max
// in the hot loop; single butterfly at the end.
// ---------------------------------------------------------------------------
__global__ __launch_bounds__(256)
void k_fusepool(const __bf16* __restrict__ cat, const float* __restrict__ W,
                const float* __restrict__ fg, const float* __restrict__ fb,
                __bf16* __restrict__ gt) {
  __shared__ __bf16 cs[64 * 512];    // [j][k]
  __shared__ __bf16 Wsh[64 * 512];   // [ol][k]
  __shared__ float gpart[8][64];
  const int lane = threadIdx.x & 31;
  const int wave = threadIdx.x >> 5;
  const int b = blockIdx.x;
  const int obase = blockIdx.y * 64;

  for (int i2 = threadIdx.x; i2 < (64 * 512) / 8; i2 += 256) {
    int lin = i2 * 8;
    const float* wp = &W[(size_t)(obase + (lin >> 9)) * 512 + (lin & 511)];
    v4f a = *(const v4f*)wp;
    v4f c = *(const v4f*)(wp + 4);
    v8bf o;
#pragma unroll
    for (int e = 0; e < 4; ++e) { o[e] = (__bf16)a[e]; o[e + 4] = (__bf16)c[e]; }
    *(v8bf*)&Wsh[lin] = o;
  }

  const int osub = wave >> 1;
  const v8f vzero = {};
  float rm[8];  // per-lane running max over this lane's columns
#pragma unroll
  for (int r = 0; r < 8; ++r) rm[r] = -1e30f;

  for (int nt = 0; nt < 32; ++nt) {
    const int n0 = nt * 64;
    __syncthreads();
    for (int i2 = threadIdx.x; i2 < (512 * 64) / 8; i2 += 256) {
      int lin = i2 * 8;
      int k = lin >> 6, j0 = lin & 63;
      v8bf val = *(const v8bf*)&cat[((size_t)b * 512 + k) * NPTS + n0 + j0];
#pragma unroll
      for (int e = 0; e < 8; ++e) cs[(j0 + e) * 512 + k] = val[e];
    }
    __syncthreads();

    v8f acc[2];
    acc[0] = vzero; acc[1] = vzero;
    for (int kc = 0; kc < 16; ++kc) {
      v16bf a = loadA(&Wsh[(osub * 16 + (lane & 15)) * 512 + kc * 32 +
                           (lane & 16)]);
#pragma unroll
      for (int t = 0; t < 2; ++t) {
        int ns = (wave & 1) + 2 * t;
        v16bf bb = loadB(&cs[(ns * 16 + (lane & 15)) * 512 + kc * 32 +
                             (lane & 16)]);
        acc[t] = wmma_bf16(a, bb, acc[t]);
      }
    }
#pragma unroll
    for (int t = 0; t < 2; ++t) {
#pragma unroll
      for (int r = 0; r < 8; ++r) {
        int o = obase + osub * 16 + r + ((lane & 16) >> 1);
        float v = fg[o] * acc[t][r] + fb[o];
        v = (v > 0.0f) ? v : 0.2f * v;
        rm[r] = fmaxf(rm[r], v);  // no cross-lane work in the hot loop
      }
    }
  }
#pragma unroll
  for (int r = 0; r < 8; ++r) rm[r] = redmax16(rm[r]);  // once at the end
  __syncthreads();
  if ((lane & 15) == 0) {
#pragma unroll
    for (int r = 0; r < 8; ++r)
      gpart[wave][osub * 16 + r + ((lane & 16) >> 1)] = rm[r];
  }
  __syncthreads();
  if (threadIdx.x < 64) {
    int o = threadIdx.x;
    int ow = (o >> 4) << 1;  // the two waves that produced this o-subtile
    float m = fmaxf(gpart[ow][o], gpart[ow + 1][o]);
    gt[(size_t)(obase + o) * 16 + b] = (__bf16)m;
  }
}

// ---------------------------------------------------------------------------
extern "C" void kernel_launch(void* const* d_in, const int* in_sizes, int n_in,
                              void* d_out, int out_size, void* d_ws, size_t ws_size,
                              hipStream_t stream) {
  (void)in_sizes; (void)n_in; (void)out_size; (void)ws_size;
  const float* x       = (const float*)d_in[0];
  const float* conv1_w = (const float*)d_in[1];
  const float* bn1_g   = (const float*)d_in[2];
  const float* bn1_b   = (const float*)d_in[3];
  const float* conv2_w = (const float*)d_in[4];
  const float* bn2_g   = (const float*)d_in[5];
  const float* bn2_b   = (const float*)d_in[6];
  const float* sa_wqk  = (const float*)d_in[7];
  const float* sa_wv   = (const float*)d_in[8];
  const float* sa_bv   = (const float*)d_in[9];
  const float* sa_wt   = (const float*)d_in[10];
  const float* sa_bt   = (const float*)d_in[11];
  const float* sa_g    = (const float*)d_in[12];
  const float* sa_b    = (const float*)d_in[13];
  const float* fuse_w  = (const float*)d_in[14];
  const float* fuse_g  = (const float*)d_in[15];
  const float* fuse_b  = (const float*)d_in[16];
  const float* lin1_w  = (const float*)d_in[17];
  const float* bn6_g   = (const float*)d_in[18];
  const float* bn6_b   = (const float*)d_in[19];
  const float* lin2_w  = (const float*)d_in[20];
  const float* lin2_b  = (const float*)d_in[21];
  const float* bn7_g   = (const float*)d_in[22];
  const float* bn7_b   = (const float*)d_in[23];
  const float* lin3_w  = (const float*)d_in[24];
  const float* lin3_b  = (const float*)d_in[25];

  char* ws = (char*)d_ws;
  size_t off = 0;
  auto alloc = [&](size_t bytes) -> void* {
    void* p = ws + off;
    off += (bytes + 255) & ~(size_t)255;
    return p;
  };
  float*  h_f  = (float*) alloc((size_t)BATCH * 128 * NPTS * 4);
  __bf16* h_bf = (__bf16*)alloc((size_t)BATCH * 128 * NPTS * 2);
  __bf16* q_bf = (__bf16*)alloc((size_t)BATCH * 32 * NPTS * 2);
  __bf16* v_bf = (__bf16*)alloc((size_t)BATCH * 128 * NPTS * 2);
  float*  rmax = (float*) alloc((size_t)BATCH * NPTS * 4);
  float*  rsiv = (float*) alloc((size_t)BATCH * NPTS * 4);
  __bf16* hmx  = (__bf16*)alloc((size_t)BATCH * 128 * NPTS * 2);
  float*  t_f  = (float*) alloc((size_t)BATCH * 128 * NPTS * 4);
  __bf16* catb = (__bf16*)alloc((size_t)BATCH * 512 * NPTS * 2);
  __bf16* gt   = (__bf16*)alloc((size_t)1024 * 16 * 2);
  __bf16* h1   = (__bf16*)alloc((size_t)512 * 16 * 2);
  __bf16* h2   = (__bf16*)alloc((size_t)256 * 16 * 2);

  k_stem<<<dim3(256), 256, 0, stream>>>(x, conv1_w, bn1_g, bn1_b,
                                        conv2_w, bn2_g, bn2_b, h_f, h_bf);

  for (int i = 0; i < 4; ++i) {
    const float* wqk = sa_wqk + (size_t)i * 32 * 128;
    const float* wv  = sa_wv  + (size_t)i * 128 * 128;
    const float* bv  = sa_bv  + (size_t)i * 128;
    const float* wt  = sa_wt  + (size_t)i * 128 * 128;
    const float* bt  = sa_bt  + (size_t)i * 128;
    const float* gg  = sa_g   + (size_t)i * 128;
    const float* bb  = sa_b   + (size_t)i * 128;

    // q = wqk @ h
    k_chanmm<<<dim3(16 * 16, 1), 256, 0, stream>>>(
        wqk, h_bf, nullptr, nullptr, nullptr, nullptr, q_bf,
        32, 128, NPTS, 16, 0, 0);
    // v = wv @ h + bv
    k_chanmm<<<dim3(16 * 16, 2), 256, 0, stream>>>(
        wv, h_bf, bv, nullptr, nullptr, nullptr, v_bf,
        128, 128, NPTS, 16, 0, 0);
    k_attn_stats<<<dim3(256), 256, 0, stream>>>(q_bf, rmax, rsiv);
    k_attn_apply<<<dim3(256), 256, 0, stream>>>(q_bf, v_bf, rmax, rsiv, h_f, hmx);
    // t = wt @ (h - x_r) + bt
    k_chanmm<<<dim3(16 * 16, 2), 256, 0, stream>>>(
        wt, hmx, bt, nullptr, nullptr, t_f, nullptr,
        128, 128, NPTS, 16, 0, 0);
    k_resid<<<dim3(16384), 256, 0, stream>>>(t_f, gg, bb, h_f, h_bf, catb, i);
  }

  k_fusepool<<<dim3(16, 16), 256, 0, stream>>>(catb, fuse_w, fuse_g, fuse_b, gt);

  // MLP head: batch mapped to the N axis (N=16, one 16-wide tile).
  k_chanmm<<<dim3(1, 8), 256, 0, stream>>>(
      lin1_w, gt, nullptr, bn6_g, bn6_b, nullptr, h1,
      512, 1024, 16, 1, 1, 0);
  k_chanmm<<<dim3(1, 4), 256, 0, stream>>>(
      lin2_w, h1, lin2_b, bn7_g, bn7_b, nullptr, h2,
      256, 512, 16, 1, 1, 0);
  k_chanmm<<<dim3(1, 1), 256, 0, stream>>>(
      lin3_w, h2, lin3_b, nullptr, nullptr, (float*)d_out, nullptr,
      40, 256, 16, 1, 0, 1);  // n-major -> d_out[b][cls]
}
